// Encoder_33998961115156
// MI455X (gfx1250) — matrix operand order repaired and checked by reading, not compiled, not run on hardware
//
#include <hip/hip_runtime.h>

#define BATCH 16
#define NPTS  4096
#define KNN   16

typedef _Float16 h16;
typedef __attribute__((ext_vector_type(16))) _Float16 v16h;
typedef __attribute__((ext_vector_type(8)))  _Float16 v8h;
typedef __attribute__((ext_vector_type(8)))  float    v8f;

#define DEVINL static __device__ __forceinline__

// ---- WMMA fragment loaders (gfx1250 16-bit layouts, ISA 7.12.2) ------------
// A 16x32 f16: lane m=lane&15; v16h elems 0..7 -> K = kf*32 + (lane>=16?8:0)+0..7
//                               elems 8..15 -> K = kf*32 + 16 + (lane>=16?8:0)+0..7
DEVINL v16h load_afrag(const h16* __restrict__ row, int kf, int lane) {
  const int base = kf * 32 + ((lane >> 4) & 1) * 8;
  union { v16h v; struct { v8h lo; v8h hi; } p; } u;
  u.p.lo = *(const v8h*)(row + base);
  u.p.hi = *(const v8h*)(row + base + 16);
  return u.v;
}
// B 32x16 f16 (col = lane&15): same K order as A fragment
DEVINL v16h load_bfrag(const h16* __restrict__ row, int kf, int lane) {
  const int base = kf * 32 + ((lane >> 4) & 1) * 8;
  union { v16h v; struct { v8h lo; v8h hi; } p; } u;
  u.p.lo = *(const v8h*)(row + base);
  u.p.hi = *(const v8h*)(row + base + 16);
  return u.v;
}
DEVINL v8f wmma_f16(v16h a, v16h b, v8f c) {
  return __builtin_amdgcn_wmma_f32_16x16x32_f16(false, a, false, b, (short)0, c,
                                                false, false);
}

// ---- sorted top-16 (ascending) insertion, all-register -----------------------
DEVINL void topk_insert(float (&bd)[16], int (&bi)[16], float e, int j) {
  if (e < bd[15]) {
    bd[15] = e; bi[15] = j;
#pragma unroll
    for (int t = 15; t > 0; --t) {
      if (bd[t] < bd[t - 1]) {
        float td = bd[t]; bd[t] = bd[t - 1]; bd[t - 1] = td;
        int   ti = bi[t]; bi[t] = bi[t - 1]; bi[t - 1] = ti;
      }
    }
  }
}

// ---- order-preserving float<->uint keys for atomic max-pool ------------------
DEVINL unsigned int fkey(float f) {
  unsigned int u = __float_as_uint(f);
  return ((int)u >= 0) ? (u | 0x80000000u) : ~u;
}
DEVINL float funkey(unsigned int k) {
  unsigned int u = (k & 0x80000000u) ? (k ^ 0x80000000u) : ~k;
  return __uint_as_float(u);
}

// ============================================================================
// K1: kNN on xyz (C=3, brute force via LDS) + local covariance features
// ============================================================================
__global__ __launch_bounds__(256) void knn3_cov_kernel(
    const float* __restrict__ x, float* __restrict__ h0) {
  __shared__ float xs[NPTS], ys[NPTS], zs[NPTS];
  const int b = blockIdx.y;
  const float* xb = x + (size_t)b * NPTS * 3;
  for (int i = threadIdx.x; i < NPTS; i += 256) {
    xs[i] = xb[i * 3 + 0]; ys[i] = xb[i * 3 + 1]; zs[i] = xb[i * 3 + 2];
  }
  __syncthreads();
  const int i = blockIdx.x * 256 + threadIdx.x;
  const float xi = xs[i], yi = ys[i], zi = zs[i];
  float bd[16]; int bi[16];
#pragma unroll
  for (int t = 0; t < 16; ++t) { bd[t] = 3.0e38f; bi[t] = 0; }
  for (int j = 0; j < NPTS; ++j) {
    const float dx = xs[j] - xi, dy = ys[j] - yi, dz = zs[j] - zi;
    topk_insert(bd, bi, dx * dx + dy * dy + dz * dz, j);
  }
  float px[16], py[16], pz[16];
  float mx = 0.f, my = 0.f, mz = 0.f;
#pragma unroll
  for (int t = 0; t < 16; ++t) {
    px[t] = xs[bi[t]]; py[t] = ys[bi[t]]; pz[t] = zs[bi[t]];
    mx += px[t]; my += py[t]; mz += pz[t];
  }
  mx *= (1.f / 16.f); my *= (1.f / 16.f); mz *= (1.f / 16.f);
  float c00 = 0, c01 = 0, c02 = 0, c11 = 0, c12 = 0, c22 = 0;
#pragma unroll
  for (int t = 0; t < 16; ++t) {
    const float dx = px[t] - mx, dy = py[t] - my, dz = pz[t] - mz;
    c00 += dx * dx; c01 += dx * dy; c02 += dx * dz;
    c11 += dy * dy; c12 += dy * dz; c22 += dz * dz;
  }
  float* o = h0 + ((size_t)b * NPTS + i) * 12;
  o[0] = xi;  o[1] = yi;  o[2] = zi;
  o[3] = c00; o[4] = c01; o[5] = c02;
  o[6] = c01; o[7] = c11; o[8] = c12;
  o[9] = c02; o[10] = c12; o[11] = c22;
}

// ============================================================================
// K2: fused BN-folded MLP 12->12->64->64 (+relu), emit f32 + f16 + sqnorm
// ============================================================================
__global__ __launch_bounds__(256) void mlp_kernel(
    const float* __restrict__ h0,
    const float* __restrict__ W1, const float* __restrict__ b1,
    const float* __restrict__ W2, const float* __restrict__ b2,
    const float* __restrict__ W3, const float* __restrict__ b3,
    float* __restrict__ h3, h16* __restrict__ h3h, float* __restrict__ sq3) {
  __shared__ float w1s[144], b1s[12], w2s[768], b2s[64], w3s[4096], b3s[64];
  const int tid = threadIdx.x;
  for (int i = tid; i < 144; i += 256)  w1s[i] = W1[i];
  for (int i = tid; i < 768; i += 256)  w2s[i] = W2[i];
  for (int i = tid; i < 4096; i += 256) w3s[i] = W3[i];
  if (tid < 12) b1s[tid] = b1[tid];
  if (tid < 64) { b2s[tid] = b2[tid]; b3s[tid] = b3[tid]; }
  __syncthreads();
  const int p = blockIdx.x * 256 + tid;
  const float* in = h0 + (size_t)p * 12;
  float a0[12];
#pragma unroll
  for (int c = 0; c < 12; ++c) a0[c] = in[c];
  float a1[12];
#pragma unroll
  for (int o = 0; o < 12; ++o) {
    float acc = b1s[o];
#pragma unroll
    for (int c = 0; c < 12; ++c) acc = fmaf(a0[c], w1s[c * 12 + o], acc);
    a1[o] = fmaxf(acc, 0.f);
  }
  float a2[64];
#pragma unroll
  for (int o = 0; o < 64; ++o) {
    float acc = b2s[o];
#pragma unroll
    for (int c = 0; c < 12; ++c) acc = fmaf(a1[c], w2s[c * 64 + o], acc);
    a2[o] = fmaxf(acc, 0.f);
  }
  float* orow = h3 + (size_t)p * 64;
  h16*  ohrow = h3h + (size_t)p * 64;
  float sq = 0.f;
  for (int o = 0; o < 64; ++o) {
    float acc = b3s[o];
#pragma unroll
    for (int c = 0; c < 64; ++c) acc = fmaf(a2[c], w3s[c * 64 + o], acc);
    const float y  = fmaxf(acc, 0.f);
    const h16   yh = (h16)y;
    const float yr = (float)yh;
    orow[o] = y; ohrow[o] = yh; sq = fmaf(yr, yr, sq);
  }
  sq3[p] = sq;
}

// ============================================================================
// K3a/K4a: WMMA-fused kNN. Rank by sq[j] - 2*<h_i,h_j>. One wave = 16 queries,
// streams 16-col candidate tiles, LDS-transposes WMMA acc, 2 lanes/query keep
// sorted top-16 lists in registers, merged at the end. B-fragments are
// double-buffered so the top-k VALU scan hides the next tile's load latency.
// ============================================================================
template <int C>
__global__ __launch_bounds__(256) void knn_wmma_kernel(
    const h16* __restrict__ feat, const float* __restrict__ sq,
    int* __restrict__ idx_out) {
  constexpr int KF = C / 32;
  __shared__ float etile[8][16][17];
  __shared__ float mdist[8][16][16];
  __shared__ int   midx[8][16][16];
  const int tid = threadIdx.x;
  const int wave = tid >> 5, lane = tid & 31;
  const int half = lane >> 4;
  const int q = lane & 15;
  const int b = blockIdx.y;
  const int qbase = (blockIdx.x * 8 + wave) * 16;
  const h16*   fb  = feat + (size_t)b * NPTS * C;
  const float* sqb = sq + (size_t)b * NPTS;

  v16h afrag[KF];
  {
    const h16* arow = fb + (size_t)(qbase + q) * C;
#pragma unroll
    for (int kf = 0; kf < KF; ++kf) afrag[kf] = load_afrag(arow, kf, lane);
  }
  float bd[16]; int bi[16];
#pragma unroll
  for (int t = 0; t < 16; ++t) { bd[t] = 3.0e38f; bi[t] = 0; }

  v16h bcur[KF];
  {
    const h16* brow = fb + (size_t)q * C;
#pragma unroll
    for (int kf = 0; kf < KF; ++kf) bcur[kf] = load_bfrag(brow, kf, lane);
  }

  for (int j0 = 0; j0 < NPTS; j0 += 16) {
    const float sqj = sqb[j0 + q];
    v8f acc;
#pragma unroll
    for (int r = 0; r < 8; ++r) acc[r] = 0.f;
#pragma unroll
    for (int kf = 0; kf < KF; ++kf) acc = wmma_f16(afrag[kf], bcur[kf], acc);

    // issue next tile's loads now; the LDS transpose + top-k scan hides them
    if (j0 + 16 < NPTS) {
      const h16* nrow = fb + (size_t)(j0 + 16 + q) * C;
#pragma unroll
      for (int kf = 0; kf < KF; ++kf) bcur[kf] = load_bfrag(nrow, kf, lane);
      __builtin_prefetch(fb + (size_t)(j0 + 64 + q) * C, 0, 1);  // global_prefetch
    }

#pragma unroll
    for (int r = 0; r < 8; ++r)
      etile[wave][r + half * 8][q] = sqj - 2.0f * acc[r];   // [M][N] (col = q)
    __asm__ __volatile__("s_wait_dscnt 0x0" ::: "memory");
#pragma unroll
    for (int c = 0; c < 8; ++c) {
      const int col = half * 8 + c;
      topk_insert(bd, bi, etile[wave][q][col], j0 + col);
    }
  }
  if (half == 1) {
#pragma unroll
    for (int t = 0; t < 16; ++t) { mdist[wave][q][t] = bd[t]; midx[wave][q][t] = bi[t]; }
  }
  __asm__ __volatile__("s_wait_dscnt 0x0" ::: "memory");
  if (half == 0) {
#pragma unroll
    for (int t = 0; t < 16; ++t) topk_insert(bd, bi, mdist[wave][q][t], midx[wave][q][t]);
    int* op = idx_out + ((size_t)b * NPTS + qbase + q) * KNN;
#pragma unroll
    for (int t = 0; t < 16; ++t) op[t] = bi[t];
  }
}

// ============================================================================
// K3b: gather-max(k=16, C=64) + composed dense 64->128 + relu; emits f32/f16/sq
// ============================================================================
__global__ __launch_bounds__(256) void gl1_kernel(
    const float* __restrict__ h3, const int* __restrict__ idx1,
    const float* __restrict__ Wg, const float* __restrict__ bg,
    float* __restrict__ h4, h16* __restrict__ h4h, float* __restrict__ sq4) {
  __shared__ float Ws[64 * 128];
  __shared__ float bs[128];
  const int tid = threadIdx.x;
  for (int i = tid; i < 64 * 128; i += 256) Ws[i] = Wg[i];
  if (tid < 128) bs[tid] = bg[tid];
  __syncthreads();
  const int p = blockIdx.x * 256 + tid;
  const int b = p >> 12;
  const size_t base = (size_t)b * NPTS;
  const int* id = idx1 + (size_t)p * KNN;
  int nb[16];
#pragma unroll
  for (int t = 0; t < 16; ++t) nb[t] = id[t];
  float m[64];
#pragma unroll
  for (int c = 0; c < 64; ++c) m[c] = -3.0e38f;
  for (int t = 0; t < 16; ++t) {
    const float* row = h3 + (base + nb[t]) * 64;
#pragma unroll
    for (int c = 0; c < 64; ++c) m[c] = fmaxf(m[c], row[c]);
  }
  float* orow = h4 + (size_t)p * 128;
  h16*  ohrow = h4h + (size_t)p * 128;
  float sq = 0.f;
  for (int ob = 0; ob < 128; ob += 16) {
    float acc[16];
#pragma unroll
    for (int t = 0; t < 16; ++t) acc[t] = bs[ob + t];
    for (int c = 0; c < 64; ++c) {
      const float mc = m[c];
#pragma unroll
      for (int t = 0; t < 16; ++t) acc[t] = fmaf(mc, Ws[c * 128 + ob + t], acc[t]);
    }
#pragma unroll
    for (int t = 0; t < 16; ++t) {
      const float y  = fmaxf(acc[t], 0.f);
      const h16   yh = (h16)y;
      const float yr = (float)yh;
      orow[ob + t] = y; ohrow[ob + t] = yh; sq = fmaf(yr, yr, sq);
    }
  }
  sq4[p] = sq;
}

// ============================================================================
// K4b1: gather-max(k=16, C=128) -> f16 rows for WMMA GEMM
// ============================================================================
__global__ __launch_bounds__(256) void gmax128_kernel(
    const float* __restrict__ h4, const int* __restrict__ idx2,
    h16* __restrict__ g2) {
  const int p = blockIdx.x * 256 + threadIdx.x;
  const int b = p >> 12;
  const size_t base = (size_t)b * NPTS;
  const int* id = idx2 + (size_t)p * KNN;
  int nb[16];
#pragma unroll
  for (int t = 0; t < 16; ++t) nb[t] = id[t];
  h16* og = g2 + (size_t)p * 128;
#pragma unroll
  for (int hblk = 0; hblk < 2; ++hblk) {
    float m[64];
#pragma unroll
    for (int c = 0; c < 64; ++c) m[c] = -3.0e38f;
    for (int t = 0; t < 16; ++t) {
      const float* row = h4 + (base + nb[t]) * 128 + hblk * 64;
#pragma unroll
      for (int c = 0; c < 64; ++c) m[c] = fmaxf(m[c], row[c]);
    }
#pragma unroll
    for (int c = 0; c < 64; ++c) og[hblk * 64 + c] = (h16)m[c];
  }
}

// ============================================================================
// K4b2: WMMA GEMM [65536x128]@[128x1024] (composed gl2) fused with global
// max-pool via keyed atomicMax. One wave = 16 rows, loops 64 N-tiles.
// Weight fragments are double-buffered; reduction+atomic hides their latency.
// ============================================================================
__global__ __launch_bounds__(256) void gemm_pool_kernel(
    const h16* __restrict__ g2, const h16* __restrict__ WT,
    const float* __restrict__ bias, unsigned int* __restrict__ pool) {
  const int wave = threadIdx.x >> 5, lane = threadIdx.x & 31;
  const int mtile = blockIdx.x * 8 + wave;          // 4096 tiles total
  const int b = mtile >> 8;                         // 256 tiles / batch
  const int row = mtile * 16 + (lane & 15);
  v16h a[4];
  {
    const h16* arow = g2 + (size_t)row * 128;
#pragma unroll
    for (int kf = 0; kf < 4; ++kf) a[kf] = load_afrag(arow, kf, lane);
  }
  unsigned int* poolb = pool + b * 1024;
  v16h bf[4];
  {
    const h16* brow = WT + (size_t)(lane & 15) * 128;
#pragma unroll
    for (int kf = 0; kf < 4; ++kf) bf[kf] = load_bfrag(brow, kf, lane);
  }
  for (int nt = 0; nt < 64; ++nt) {
    const int n = nt * 16 + (lane & 15);
    const float bv = bias[n];
    v8f acc;
#pragma unroll
    for (int r = 0; r < 8; ++r) acc[r] = 0.f;
#pragma unroll
    for (int kf = 0; kf < 4; ++kf) acc = wmma_f16(a[kf], bf[kf], acc);

    if (nt + 1 < 64) {  // prefetch next N-tile's weights
      const h16* brow = WT + (size_t)(n + 16) * 128;
#pragma unroll
      for (int kf = 0; kf < 4; ++kf) bf[kf] = load_bfrag(brow, kf, lane);
    }

    float mv = acc[0] + bv;
#pragma unroll
    for (int r = 1; r < 8; ++r) mv = fmaxf(mv, acc[r] + bv);
    mv = fmaxf(mv, __shfl_xor(mv, 16, 32));         // combine M=0..7 | 8..15
    if (lane < 16) atomicMax(poolb + n, fkey(mv));
  }
}

// ============================================================================
// Head: per-batch pooled[1024] -> relu(dense 1024) -> dense 512 -> d_out
// ============================================================================
__global__ __launch_bounds__(256) void head_kernel(
    const unsigned int* __restrict__ pool,
    const float* __restrict__ W4, const float* __restrict__ b4,
    const float* __restrict__ W5, const float* __restrict__ b5,
    float* __restrict__ out) {
  __shared__ float p[1024];
  __shared__ float t4[1024];
  const int b = blockIdx.x, tid = threadIdx.x;
  for (int c = tid; c < 1024; c += 256) p[c] = funkey(pool[b * 1024 + c]);
  __syncthreads();
  for (int o = tid; o < 1024; o += 256) {
    float acc = b4[o];
    for (int c = 0; c < 1024; ++c) acc = fmaf(p[c], W4[c * 1024 + o], acc);
    t4[o] = fmaxf(acc, 0.f);
  }
  __syncthreads();
  for (int o = tid; o < 512; o += 256) {
    float acc = b5[o];
    for (int c = 0; c < 1024; ++c) acc = fmaf(t4[c], W5[c * 512 + o], acc);
    out[b * 512 + o] = acc;
  }
}

// ============================================================================
// Prep kernels: BN folding, graph-layer weight composition, pool init
// ============================================================================
__global__ void fold_bn_kernel(const float* __restrict__ w, const float* __restrict__ b,
                               const float* __restrict__ gamma, const float* __restrict__ beta,
                               const float* __restrict__ mean, const float* __restrict__ var,
                               float* __restrict__ wf, float* __restrict__ bf,
                               int cin, int cout) {
  const int i = blockIdx.x * 256 + threadIdx.x;
  if (i < cin * cout) {
    const int o = i % cout;
    const float s = gamma[o] * rsqrtf(var[o] + 1e-3f);
    wf[i] = w[i] * s;
  }
  if (i < cout) {
    const float s = gamma[i] * rsqrtf(var[i] + 1e-3f);
    bf[i] = (b[i] - mean[i]) * s + beta[i];
  }
}

__global__ void compose_gl1_kernel(const float* __restrict__ Aw, const float* __restrict__ Ab,
                                   const float* __restrict__ Bw, const float* __restrict__ Bb,
                                   float* __restrict__ W, float* __restrict__ bias) {
  // A:64x64, B:64x128 -> W:64x128 row-major f32
  const int i = blockIdx.x * 256 + threadIdx.x;
  const int ci = i >> 7, co = i & 127;
  float acc = 0.f;
  for (int m = 0; m < 64; ++m) acc = fmaf(Aw[ci * 64 + m], Bw[m * 128 + co], acc);
  W[i] = acc;
  if (ci == 0) {
    float bacc = Bb[co];
    for (int m = 0; m < 64; ++m) bacc = fmaf(Ab[m], Bw[m * 128 + co], bacc);
    bias[co] = bacc;
  }
}

__global__ void compose_gl2_kernel(const float* __restrict__ Aw, const float* __restrict__ Ab,
                                   const float* __restrict__ Bw, const float* __restrict__ Bb,
                                   h16* __restrict__ WT, float* __restrict__ bias) {
  // A:128x128, B:128x1024 -> WT[o][c] f16 (transposed, 1024x128), bias[1024]
  const int i = blockIdx.x * 256 + threadIdx.x;
  const int o = i >> 7, c = i & 127;
  float acc = 0.f;
  for (int m = 0; m < 128; ++m) acc = fmaf(Aw[c * 128 + m], Bw[m * 1024 + o], acc);
  WT[(size_t)o * 128 + c] = (h16)acc;
  if (c == 0) {
    float bacc = Bb[o];
    for (int m = 0; m < 128; ++m) bacc = fmaf(Ab[m], Bw[m * 1024 + o], bacc);
    bias[o] = bacc;
  }
}

__global__ void init_pool_kernel(unsigned int* __restrict__ pool) {
  const int i = blockIdx.x * 256 + threadIdx.x;
  if (i < BATCH * 1024) pool[i] = 0u;   // key 0 < key(any finite float)
}

// ============================================================================
extern "C" void kernel_launch(void* const* d_in, const int* in_sizes, int n_in,
                              void* d_out, int out_size, void* d_ws, size_t ws_size,
                              hipStream_t stream) {
  const float *x, *c1w, *c1b, *bn1g, *bn1b, *bn1m, *bn1v,
      *c2w, *c2b, *bn2g, *bn2b, *bn2m, *bn2v,
      *c3w, *c3b, *bn3g, *bn3b, *bn3m, *bn3v,
      *g1lw, *g1lb, *g1cw, *g1cb, *g2lw, *g2lb, *g2cw, *g2cb,
      *c4w, *c4b, *c5w, *c5b;
  auto F = [&](int i) { return (const float*)d_in[i]; };
  if (in_sizes[0] == BATCH * NPTS * 3) {
    // dict insertion order: x, then params depth-first
    x = F(0);
    c1w = F(1);  c1b = F(2);  bn1g = F(3);  bn1b = F(4);  bn1m = F(5);  bn1v = F(6);
    c2w = F(7);  c2b = F(8);  bn2g = F(9);  bn2b = F(10); bn2m = F(11); bn2v = F(12);
    c3w = F(13); c3b = F(14); bn3g = F(15); bn3b = F(16); bn3m = F(17); bn3v = F(18);
    g1lw = F(19); g1lb = F(20); g1cw = F(21); g1cb = F(22);
    g2lw = F(23); g2lb = F(24); g2cw = F(25); g2cb = F(26);
    c4w = F(27); c4b = F(28); c5w = F(29); c5b = F(30);
  } else {
    // jax tree-sorted order: params (keys sorted, leaves sorted) then x
    bn1b = F(0);  bn1g = F(1);  bn1m = F(2);  bn1v = F(3);
    bn2b = F(4);  bn2g = F(5);  bn2m = F(6);  bn2v = F(7);
    bn3b = F(8);  bn3g = F(9);  bn3m = F(10); bn3v = F(11);
    c1b = F(12); c1w = F(13); c2b = F(14); c2w = F(15); c3b = F(16); c3w = F(17);
    c4b = F(18); c4w = F(19); c5b = F(20); c5w = F(21);
    g1cb = F(22); g1cw = F(23); g1lb = F(24); g1lw = F(25);
    g2cb = F(26); g2cw = F(27); g2lb = F(28); g2lw = F(29);
    x = F(30);
  }
  (void)n_in; (void)out_size; (void)ws_size;

  char* ws = (char*)d_ws;
  size_t off = 0;
  auto carve = [&](size_t bytes) -> void* {
    void* p = ws + off;
    off = (off + bytes + 255) & ~(size_t)255;
    return p;
  };
  float* W1f = (float*)carve(144 * 4);            float* b1f = (float*)carve(12 * 4);
  float* W2f = (float*)carve(768 * 4);            float* b2f = (float*)carve(64 * 4);
  float* W3f = (float*)carve(4096 * 4);           float* b3f = (float*)carve(64 * 4);
  float* Wg1 = (float*)carve(8192 * 4);           float* bg1 = (float*)carve(128 * 4);
  h16* Wg2T  = (h16*)carve((size_t)131072 * 2);   float* bg2 = (float*)carve(1024 * 4);
  float* h0  = (float*)carve((size_t)BATCH * NPTS * 12 * 4);
  float* h3  = (float*)carve((size_t)BATCH * NPTS * 64 * 4);
  h16*   h3h = (h16*)carve((size_t)BATCH * NPTS * 64 * 2);
  float* sq3 = (float*)carve((size_t)BATCH * NPTS * 4);
  int*   idx1 = (int*)carve((size_t)BATCH * NPTS * KNN * 4);
  float* h4  = (float*)carve((size_t)BATCH * NPTS * 128 * 4);
  h16*   h4h = (h16*)carve((size_t)BATCH * NPTS * 128 * 2);
  float* sq4 = (float*)carve((size_t)BATCH * NPTS * 4);
  int*   idx2 = (int*)carve((size_t)BATCH * NPTS * KNN * 4);
  h16*   g2  = (h16*)carve((size_t)BATCH * NPTS * 128 * 2);
  unsigned int* pool = (unsigned int*)carve((size_t)BATCH * 1024 * 4);

  fold_bn_kernel<<<1, 256, 0, stream>>>(c1w, c1b, bn1g, bn1b, bn1m, bn1v, W1f, b1f, 12, 12);
  fold_bn_kernel<<<3, 256, 0, stream>>>(c2w, c2b, bn2g, bn2b, bn2m, bn2v, W2f, b2f, 12, 64);
  fold_bn_kernel<<<16, 256, 0, stream>>>(c3w, c3b, bn3g, bn3b, bn3m, bn3v, W3f, b3f, 64, 64);
  compose_gl1_kernel<<<32, 256, 0, stream>>>(g1lw, g1lb, g1cw, g1cb, Wg1, bg1);
  compose_gl2_kernel<<<512, 256, 0, stream>>>(g2lw, g2lb, g2cw, g2cb, Wg2T, bg2);
  init_pool_kernel<<<64, 256, 0, stream>>>(pool);

  knn3_cov_kernel<<<dim3(16, 16), 256, 0, stream>>>(x, h0);
  mlp_kernel<<<256, 256, 0, stream>>>(h0, W1f, b1f, W2f, b2f, W3f, b3f, h3, h3h, sq3);
  knn_wmma_kernel<64><<<dim3(32, 16), 256, 0, stream>>>(h3h, sq3, idx1);
  gl1_kernel<<<256, 256, 0, stream>>>(h3, idx1, Wg1, bg1, h4, h4h, sq4);
  knn_wmma_kernel<128><<<dim3(32, 16), 256, 0, stream>>>(h4h, sq4, idx2);
  gmax128_kernel<<<256, 256, 0, stream>>>(h4, idx2, g2);
  gemm_pool_kernel<<<512, 256, 0, stream>>>(g2, Wg2T, bg2, pool);
  head_kernel<<<16, 256, 0, stream>>>(pool, c4w, c4b, c5w, c5b, (float*)d_out);
}
